// ProductLayer_54065048322685
// MI455X (gfx1250) — compile-verified, matrix operand order checked by reading, not run on hardware
//
#include <hip/hip_runtime.h>
#include <hip/hip_bf16.h>

// ---------------------------------------------------------------------------
// ProductLayer fused pipeline for MI455X (gfx1250, wave32, WMMA + TDM).
//   B=1024, N=24, C_IN=C_OUT=2, E=256, H=4, DH=64
// Pipeline:
//   1) convert feat -> bf16; transpose+convert Wq/Wk/Wv and 5D kernel -> bf16
//   2) attention kernel: TDM-staged X, QKV via v_wmma_f32_16x16x32_bf16,
//      VALU softmax (seq len 24), ctx -> fp32 workspace
//   3) pk kernel: TDM-staged A/ctx tiles, 96 GEMMs (1024x256x256) via WMMA,
//      fused ctx scaling, bias, residual, layernorm epilogue
// ---------------------------------------------------------------------------

typedef __attribute__((ext_vector_type(16))) __bf16 v16bf;
typedef __attribute__((ext_vector_type(8)))  float  v8f;
typedef __attribute__((ext_vector_type(4)))  unsigned int v4u;
typedef __attribute__((ext_vector_type(8)))  int    v8i_;
typedef __attribute__((ext_vector_type(4)))  int    v4i_;

#define B_    1024
#define N_    24
#define CIN_  2
#define COUT_ 2
#define E_    256
#define H_    4
#define DH_   64

// ---------------------------------------------------------------------------
// Tensor Data Mover: 2D tile load global -> LDS (CDNA5 ISA ch.8, D# layout).
//   lds_off          : LDS byte address of tile start
//   gaddr            : global byte address of tile start
//   dsz              : data_size code (0=1B,1=2B,2=4B,3=8B)
//   width            : elements per row (tensor_dim0 == tile_dim0)
//   rows             : rows in tile (tensor_dim1 == tile_dim1)
//   row_stride       : elements between consecutive rows (tensor_dim0_stride)
// ---------------------------------------------------------------------------
__device__ __forceinline__ void tdm_load_2d(unsigned lds_off,
                                            unsigned long long gaddr,
                                            unsigned dsz, unsigned width,
                                            unsigned rows,
                                            unsigned long long row_stride) {
    v4u g0;
    g0[0] = 1u;                                                  // count=1, user mode
    g0[1] = lds_off;                                             // lds_addr
    g0[2] = (unsigned)(gaddr & 0xFFFFFFFFull);                   // global_addr lo
    g0[3] = (unsigned)((gaddr >> 32) & 0x1FFFFFFull) | (2u << 30); // addr hi | type=2

    v8i_ g1;
    g1[0] = (int)(dsz << 16);                                    // wg_mask=0, data_size
    g1[1] = (int)((width & 0xFFFFu) << 16);                      // tensor_dim0[15:0]
    g1[2] = (int)((width >> 16) | ((rows & 0xFFFFu) << 16));     // dim0 hi | tensor_dim1 lo
    g1[3] = (int)((rows >> 16) | ((width & 0xFFFFu) << 16));     // dim1 hi | tile_dim0
    g1[4] = (int)(rows & 0xFFFFu);                               // tile_dim1 (tile_dim2=0)
    g1[5] = (int)(row_stride & 0xFFFFFFFFull);                   // dim0_stride lo
    g1[6] = (int)((row_stride >> 32) & 0xFFFFull);               // dim0_stride hi (dim1_stride=0)
    g1[7] = 0;

    v4i_ gz = {0, 0, 0, 0};
#if defined(__clang_major__) && (__clang_major__ >= 23)
    v8i_ gz8 = {0, 0, 0, 0, 0, 0, 0, 0};
    __builtin_amdgcn_tensor_load_to_lds(g0, g1, gz, gz, gz8, 0);
#else
    __builtin_amdgcn_tensor_load_to_lds(g0, g1, gz, gz, 0);
#endif
}

// ---------------------------------------------------------------------------
// WMMA 16x16x32 bf16 fragment loader.
// Layout (CDNA5 ISA 7.12.2, 16-bit A 16x32 / B 32x16, symmetric):
//   lanes 0-15 : row/col = lane,     K = k0+{0..7} U k0+{16..23}
//   lanes 16-31: row/col = lane-16,  K = k0+{8..15} U k0+{24..31}
// Source matrix layout is [rowcol][k], row-major, stride E_ = 256.
// ---------------------------------------------------------------------------
__device__ __forceinline__ v16bf load_frag16(const __bf16* base, int rc0, int k0, int lane) {
    const __bf16* p = base + (rc0 + (lane & 15)) * E_ + k0 + ((lane >> 4) << 3);
    v16bf f;
#pragma unroll
    for (int j = 0; j < 8; ++j) {
        f[j]     = p[j];
        f[j + 8] = p[j + 16];
    }
    return f;
}

// ---------------------------------------------------------------------------
// Converters
// ---------------------------------------------------------------------------
__global__ void k_cvt_feat(const float* __restrict__ in, __bf16* __restrict__ out, int n) {
    int i = blockIdx.x * blockDim.x + threadIdx.x;
    if (i < n) out[i] = (__bf16)in[i];
}

// wt[f][e] = w[e][f]  (256x256), coalesced writes
__global__ void k_cvt_wT(const float* __restrict__ w, __bf16* __restrict__ wt) {
    int idx = blockIdx.x * blockDim.x + threadIdx.x;
    int f = idx >> 8, e = idx & 255;
    wt[idx] = (__bf16)w[e * E_ + f];
}

// kt[o,i,n][f][e] = kernel[o,i,n][e][f]  (96 matrices of 256x256)
__global__ void k_cvt_kT(const float* __restrict__ k, __bf16* __restrict__ kt) {
    int idx = blockIdx.x * blockDim.x + threadIdx.x;
    int mat = idx >> 16, f = (idx >> 8) & 255, e = idx & 255;
    kt[idx] = (__bf16)k[mat * (E_ * E_) + e * E_ + f];
}

// ---------------------------------------------------------------------------
// Attention: one block (8 waves) per m in [0, B*CIN).
//   X[n][e] = feat[b, n, i, e]   (b=m>>1, i=m&1), rows 24..31 zero-padded
//   Q/K/V = X @ W^T-layout weights via WMMA, then VALU attention (seq 24).
//   ctx out: [m][n][e] fp32
// ---------------------------------------------------------------------------
__global__ void __launch_bounds__(256) k_attn(
    const __bf16* __restrict__ featb,
    const __bf16* __restrict__ wqT, const __bf16* __restrict__ wkT, const __bf16* __restrict__ wvT,
    const float* __restrict__ bq, const float* __restrict__ bk, const float* __restrict__ bv,
    float* __restrict__ ctx)
{
    extern __shared__ char smem[];
    __bf16* xs  = (__bf16*)smem;                 // [32][256] bf16
    __bf16* qkv = xs + 32 * E_;                  // 3 x [32][256] bf16 (Q,K,V)
    float*  sc  = (float*)(qkv + 3 * 32 * E_);   // [H][24][24] fp32

    const int m = blockIdx.x;
    const int b = m >> 1, ich = m & 1;
    const int tid = threadIdx.x, lane = tid & 31, wave = tid >> 5;

    // ---- zero-fill pad rows 24..31 (disjoint from TDM target rows) ----
    {
        int row = 24 + (tid >> 5);
        int c8  = (tid & 31) << 3;
        *(uint4*)(xs + row * E_ + c8) = make_uint4(0u, 0u, 0u, 0u);
    }
    // ---- TDM: stage X rows 0..23 into LDS ----
    if (wave == 0) {
        unsigned lds_off = (unsigned)(uintptr_t)xs;
        unsigned long long ga = (unsigned long long)(uintptr_t)
            (featb + ((size_t)((b * N_) * CIN_ + ich)) * E_);
        tdm_load_2d(lds_off, ga, 1u, E_, N_, (unsigned long long)(CIN_ * E_));
        __builtin_amdgcn_s_wait_tensorcnt(0);
    }
    __syncthreads();

    // ---- QKV projection: 96 output tiles (3 mats x 2 Mtiles x 16 Ntiles) ----
    __builtin_prefetch(wqT, 0, 2);
    __builtin_prefetch(wkT, 0, 2);
    __builtin_prefetch(wvT, 0, 2);
    for (int t = wave * 12; t < wave * 12 + 12; ++t) {
        int mat = t / 32;
        int r   = t % 32;
        int mt  = r & 1;        // M tile (rows 0-15 / 16-31)
        int nt  = r >> 1;       // N tile (cols)
        const __bf16* wt = (mat == 0) ? wqT : (mat == 1 ? wkT : wvT);
        int col = nt * 16 + (lane & 15);
        float bias = (mat == 0) ? bq[col] : (mat == 1 ? bk[col] : bv[col]);
        v8f acc;
#pragma unroll
        for (int j = 0; j < 8; ++j) acc[j] = bias;
#pragma unroll
        for (int k0 = 0; k0 < E_; k0 += 32) {
            v16bf a  = load_frag16(xs, mt * 16, k0, lane);
            v16bf bb = load_frag16(wt, nt * 16, k0, lane);
            acc = __builtin_amdgcn_wmma_f32_16x16x32_bf16(
                false, a, false, bb, (short)0, acc, false, false);
        }
        // D layout: VGPR j -> M = mt*16 + j + 8*(lane>=16); N = col
        __bf16* dst = qkv + mat * 32 * E_;
        int half = (lane >> 4) << 3;
#pragma unroll
        for (int j = 0; j < 8; ++j) {
            int row = mt * 16 + j + half;
            dst[row * E_ + col] = (__bf16)acc[j];
        }
    }
    __syncthreads();

    const __bf16* qsm = qkv;
    const __bf16* ksm = qkv + 32 * E_;
    const __bf16* vsm = qkv + 64 * E_;

    // ---- scores: (H*24*24) dot products of length 64, scale 1/sqrt(64) ----
    for (int idx = tid; idx < H_ * N_ * N_; idx += 256) {
        int h   = idx / (N_ * N_);
        int rem = idx - h * (N_ * N_);
        int qn  = rem / N_, kn = rem % N_;
        float s = 0.f;
#pragma unroll 8
        for (int d = 0; d < DH_; ++d)
            s += (float)qsm[qn * E_ + h * DH_ + d] * (float)ksm[kn * E_ + h * DH_ + d];
        sc[(h * N_ + qn) * N_ + kn] = s * 0.125f;
    }
    __syncthreads();

    // ---- softmax over 24 keys (96 rows) ----
    if (tid < H_ * N_) {
        float* row = sc + tid * N_;
        float mx = row[0];
#pragma unroll
        for (int k = 1; k < N_; ++k) mx = fmaxf(mx, row[k]);
        float sum = 0.f;
#pragma unroll
        for (int k = 0; k < N_; ++k) { float e = __expf(row[k] - mx); row[k] = e; sum += e; }
        float inv = 1.f / sum;
#pragma unroll
        for (int k = 0; k < N_; ++k) row[k] *= inv;
    }
    __syncthreads();

    // ---- ctx[qn][e] = sum_k probs * V ----
    for (int idx = tid; idx < N_ * E_; idx += 256) {
        int qn = idx >> 8, e = idx & 255;
        int h  = e >> 6;
        float s = 0.f;
#pragma unroll
        for (int kn = 0; kn < N_; ++kn)
            s += sc[(h * N_ + qn) * N_ + kn] * (float)vsm[kn * E_ + e];
        ctx[((size_t)m * N_ + qn) * E_ + e] = s;
    }
}

// ---------------------------------------------------------------------------
// Main kernel: grid (B/32, N, COUT). Per block: 32 b-rows x 256 f-cols.
//   for i in {0,1}: pk = A_i (32x256) @ kT[o,i,n] (256x256) via WMMA,
//                   pkq += pk * ctx[b,i,n,f]
//   pkq += bias[n,o] + feat[b,n,o,f]; layernorm over f; store.
// ---------------------------------------------------------------------------
__global__ void __launch_bounds__(256) k_pk(
    const __bf16* __restrict__ featb, const __bf16* __restrict__ kT,
    const float* __restrict__ ctx,   const float* __restrict__ feat,
    const float* __restrict__ bias2, const float* __restrict__ gamma,
    const float* __restrict__ beta,  float* __restrict__ out)
{
    extern __shared__ char smem[];
    __bf16* as   = (__bf16*)smem;                // [2][32][256] bf16
    float*  cs   = (float*)(as + 2 * 32 * E_);   // [2][32][256] fp32
    float*  rsum = cs + 2 * 32 * E_;             // [32]
    float*  rssq = rsum + 32;                    // [32]

    const int bblk = blockIdx.x, n = blockIdx.y, o = blockIdx.z;
    const int b0 = bblk * 32;
    const int tid = threadIdx.x, lane = tid & 31, wave = tid >> 5;

    // ---- TDM: stage A (both i channels) and ctx rows into LDS ----
    if (wave == 0) {
#pragma unroll
        for (int i = 0; i < CIN_; ++i) {
            // A_i: 32 rows x 256 bf16, row stride (delta b) = N_*CIN_*E_ elems
            unsigned lds_a = (unsigned)(uintptr_t)(as + i * 32 * E_);
            unsigned long long ga = (unsigned long long)(uintptr_t)
                (featb + ((size_t)((b0 * N_ + n) * CIN_ + i)) * E_);
            tdm_load_2d(lds_a, ga, 1u, E_, 32u,
                        (unsigned long long)(N_ * CIN_ * E_));
            // ctx_i: 32 rows x 256 fp32, row stride (delta b) = CIN_*N_*E_ elems
            unsigned lds_c = (unsigned)(uintptr_t)(cs + i * 32 * E_);
            unsigned long long gc = (unsigned long long)(uintptr_t)
                (ctx + ((size_t)((b0 * CIN_ + i) * N_ + n)) * E_);
            tdm_load_2d(lds_c, gc, 2u, E_, 32u,
                        (unsigned long long)(CIN_ * N_ * E_));
        }
        __builtin_amdgcn_s_wait_tensorcnt(0);
    }
    if (tid < 32) { rsum[tid] = 0.f; rssq[tid] = 0.f; }
    __syncthreads();

    const int mt   = wave & 1;       // M tile
    const int ntb  = wave >> 1;      // base N tile; owns ntb + {0,4,8,12}
    const int colq = lane & 15;
    const int half = (lane >> 4) << 3;

    v8f pkq[4];
#pragma unroll
    for (int t = 0; t < 4; ++t)
#pragma unroll
        for (int j = 0; j < 8; ++j) pkq[t][j] = 0.f;

#pragma unroll
    for (int i = 0; i < CIN_; ++i) {
        const __bf16* ai = as + i * 32 * E_;
        const __bf16* kb = kT + ((size_t)((o * CIN_ + i) * N_ + n)) * (E_ * E_); // [f][e]
        __builtin_prefetch(kb, 0, 1);
        v8f acc[4];
#pragma unroll
        for (int t = 0; t < 4; ++t)
#pragma unroll
            for (int j = 0; j < 8; ++j) acc[t][j] = 0.f;
#pragma unroll
        for (int k0 = 0; k0 < E_; k0 += 32) {
            v16bf a = load_frag16(ai, mt * 16, k0, lane);
#pragma unroll
            for (int t = 0; t < 4; ++t) {
                int nt = ntb + t * 4;
                v16bf bb = load_frag16(kb, nt * 16, k0, lane);
                acc[t] = __builtin_amdgcn_wmma_f32_16x16x32_bf16(
                    false, a, false, bb, (short)0, acc[t], false, false);
            }
        }
        // elementwise ctx scaling before summing over i
        const float* ci = cs + i * 32 * E_;
#pragma unroll
        for (int t = 0; t < 4; ++t) {
            int col = (ntb + t * 4) * 16 + colq;
#pragma unroll
            for (int j = 0; j < 8; ++j) {
                int row = mt * 16 + j + half;
                pkq[t][j] += acc[t][j] * ci[row * E_ + col];
            }
        }
    }

    // ---- bias + residual + layernorm partial sums ----
    float bval = bias2[n * COUT_ + o];
#pragma unroll
    for (int t = 0; t < 4; ++t) {
        int col = (ntb + t * 4) * 16 + colq;
#pragma unroll
        for (int j = 0; j < 8; ++j) {
            int row = mt * 16 + j + half;
            float resid =
                feat[((size_t)(((b0 + row) * N_ + n) * CIN_ + o)) * E_ + col];
            float v = pkq[t][j] + bval + resid;
            pkq[t][j] = v;
            float s = v, s2 = v * v;
#pragma unroll
            for (int msk = 8; msk >= 1; msk >>= 1) {
                s  += __shfl_xor(s, msk, 16);
                s2 += __shfl_xor(s2, msk, 16);
            }
            if (colq == 0) {
                atomicAdd(&rsum[row], s);
                atomicAdd(&rssq[row], s2);
            }
        }
    }
    __syncthreads();
    if (tid < 32) {
        float mu  = rsum[tid] * (1.f / (float)E_);
        float var = rssq[tid] * (1.f / (float)E_) - mu * mu;
        rsum[tid] = mu;
        rssq[tid] = rsqrtf(var + 1e-12f);
    }
    __syncthreads();

    // ---- normalize + store ----
#pragma unroll
    for (int t = 0; t < 4; ++t) {
        int col = (ntb + t * 4) * 16 + colq;
        float g = gamma[col], be = beta[col];
#pragma unroll
        for (int j = 0; j < 8; ++j) {
            int row = mt * 16 + j + half;
            float v = (pkq[t][j] - rsum[row]) * rssq[row] * g + be;
            out[((size_t)(((b0 + row) * N_ + n) * COUT_ + o)) * E_ + col] = v;
        }
    }
}

// ---------------------------------------------------------------------------
// Launcher
// ---------------------------------------------------------------------------
extern "C" void kernel_launch(void* const* d_in, const int* in_sizes, int n_in,
                              void* d_out, int out_size, void* d_ws, size_t ws_size,
                              hipStream_t stream) {
    (void)in_sizes; (void)n_in; (void)out_size; (void)ws_size;

    const float* feat  = (const float*)d_in[0];
    const float* kern  = (const float*)d_in[1];
    const float* bias  = (const float*)d_in[2];
    const float* Wq    = (const float*)d_in[3];
    const float* bq    = (const float*)d_in[4];
    const float* Wk    = (const float*)d_in[5];
    const float* bk    = (const float*)d_in[6];
    const float* Wv    = (const float*)d_in[7];
    const float* bv    = (const float*)d_in[8];
    const float* gamma = (const float*)d_in[9];
    const float* beta  = (const float*)d_in[10];
    float* out = (float*)d_out;

    // workspace partition (all 16B aligned)
    __bf16* featb = (__bf16*)d_ws;                                   // 12,582,912 bf16
    __bf16* kT    = featb + (size_t)B_ * N_ * CIN_ * E_;             //  6,291,456 bf16
    __bf16* wqT   = kT + (size_t)COUT_ * CIN_ * N_ * E_ * E_;        //     65,536 bf16
    __bf16* wkT   = wqT + E_ * E_;
    __bf16* wvT   = wkT + E_ * E_;
    float*  ctx   = (float*)(wvT + E_ * E_);                         // 12,582,912 fp32

    const int nf = B_ * N_ * CIN_ * E_;
    k_cvt_feat<<<nf / 256, 256, 0, stream>>>(feat, featb, nf);
    k_cvt_wT<<<(E_ * E_) / 256, 256, 0, stream>>>(Wq, wqT);
    k_cvt_wT<<<(E_ * E_) / 256, 256, 0, stream>>>(Wk, wkT);
    k_cvt_wT<<<(E_ * E_) / 256, 256, 0, stream>>>(Wv, wvT);
    const int nk = COUT_ * CIN_ * N_ * E_ * E_;
    k_cvt_kT<<<nk / 256, 256, 0, stream>>>(kern, kT);

    const size_t sh_attn = (size_t)(4 * 32 * E_) * sizeof(__bf16)
                         + (size_t)(H_ * N_ * N_) * sizeof(float);   // 74,752 B
    k_attn<<<B_ * CIN_, 256, sh_attn, stream>>>(featb, wqT, wkT, wvT, bq, bk, bv, ctx);

    const size_t sh_pk = (size_t)(2 * 32 * E_) * sizeof(__bf16)
                       + (size_t)(2 * 32 * E_ + 64) * sizeof(float); // 98,560 B
    k_pk<<<dim3(B_ / 32, N_, COUT_), 256, sh_pk, stream>>>(
        featb, kT, ctx, feat, bias, gamma, beta, out);
}
